// CausalSelfAttention_46926812676761
// MI455X (gfx1250) — compile-verified, matrix operand order checked by reading, not run on hardware
//
#include <hip/hip_runtime.h>
#include <hip/hip_bf16.h>

typedef _Float16 half_t;
typedef __attribute__((ext_vector_type(16))) _Float16 v16h;
typedef __attribute__((ext_vector_type(8)))  _Float16 v8h;
typedef __attribute__((ext_vector_type(8)))  float    v8f;
typedef __attribute__((ext_vector_type(4)))  unsigned int u32x4;
typedef __attribute__((ext_vector_type(4)))  int i32x4;
typedef __attribute__((ext_vector_type(8)))  int i32x8;

#define D_MODEL 1024
#define N_HEADS 16
#define HEAD_DIM 64
#define SEQ 2048
#define BATCH 4
#define MROWS (BATCH * SEQ)   // 8192
#define N_QKV (3 * D_MODEL)   // 3072

static __device__ __forceinline__ v8f wmma_f16(v16h a, v16h b, v8f c) {
  return __builtin_amdgcn_wmma_f32_16x16x32_f16(false, a, false, b, (short)0, c,
                                                false, false);
}

// Load a 16x32 f16 A-fragment (or 32x16 B-fragment whose "rows" are the
// 16-wide dimension) from a row-major array of halves.
// CDNA5 16-bit A 16x32 layout:
//   lanes 0-15 : halves 0..7 = K 0..7,  halves 8..15 = K 16..23
//   lanes 16-31: halves 0..7 = K 8..15, halves 8..15 = K 24..31
// strideH must be a multiple of 8 halves so the two b128 reads stay aligned.
static __device__ __forceinline__ v16h load_frag(const half_t* base, int strideH,
                                                 int lane) {
  const int r  = lane & 15;
  const int bk = (lane & 16) ? 8 : 0;
  const half_t* p = base + r * strideH + bk;
  v8h lo = *(const v8h*)(p);
  v8h hi = *(const v8h*)(p + 16);
  v16h out;
#pragma unroll
  for (int j = 0; j < 8; ++j) { out[j] = lo[j]; out[j + 8] = hi[j]; }
  return out;
}

// ---------------------------------------------------------------------------
// Tensor Data Mover: load a [32 x 64] f16 tile (row stride 64 halves in
// memory) into LDS with 8 halves of padding per row (LDS row stride 72).
// D# built per CDNA5 ISA ch.8 (group0: count/lds/global/type, group1: sizes).
// ---------------------------------------------------------------------------
static __device__ __forceinline__ void tdm_load_k32x64(const half_t* gsrc,
                                                       unsigned lds_byte_off) {
  const unsigned long long ga = (unsigned long long)(size_t)gsrc;
  u32x4 g0;
  g0[0] = 1u;                                   // count=1, user descriptor
  g0[1] = lds_byte_off;                         // lds_addr
  g0[2] = (unsigned)(ga & 0xFFFFFFFFu);         // global_addr[31:0]
  g0[3] = (unsigned)((ga >> 32) & 0x01FFFFFFu)  // global_addr[56:32]
        | (2u << 30);                           // type=2 ("image")
  i32x8 g1;
  g1[0] = (1 << 16)        // data_size = 2 bytes
        | (1 << 20)        // pad_enable
        | (4 << 22)        // pad_interval code 4 -> every 32 dwords (one row)
        | (3 << 25);       // pad_amount  code 3 -> 4 dwords (8 halves)
  g1[1] = (HEAD_DIM & 0xFFFF) << 16;  // tensor_dim0 = 64  (bits 79:48)
  g1[2] = (SEQ & 0xFFFF) << 16;       // tensor_dim1 = 2048 (bits 111:80)
  g1[3] = HEAD_DIM << 16;             // tile_dim0 = 64    (bits 127:112)
  g1[4] = 32;                         // tile_dim1 = 32    (bits 143:128)
  g1[5] = HEAD_DIM;                   // tensor_dim0_stride = 64 (bits 207:160)
  g1[6] = 0;
  g1[7] = 0;
  const i32x4 z4 = {0, 0, 0, 0};
#if defined(__clang_major__) && (__clang_major__ >= 23)
  const i32x8 z8 = {0, 0, 0, 0, 0, 0, 0, 0};
  __builtin_amdgcn_tensor_load_to_lds(g0, g1, z4, z4, z8, 0);
#else
  __builtin_amdgcn_tensor_load_to_lds(g0, g1, z4, z4, 0);
#endif
}

// ---------------------------------------------------------------------------
// Kernel 1: QKV projection. C[8192,3072] = X[8192,1024] @ W[1024,3072] + bias
// Block tile 128x128, BK=64; 8 waves, each wave owns a 32x64 output tile
// (8 f32 accumulators -> 16 WMMAs per K-step per wave).
// Results scattered as f16 into q/k/v buffers laid out [b][h][t][hd].
// ---------------------------------------------------------------------------
__global__ __launch_bounds__(256) void qkv_gemm_kernel(
    const float* __restrict__ x, const float* __restrict__ w,
    const float* __restrict__ bias, half_t* __restrict__ qf,
    half_t* __restrict__ kf, half_t* __restrict__ vf) {
  __shared__ half_t Xs[128 * 72];   // [m][k] rows, stride 72
  __shared__ half_t Wst[128 * 72];  // [n][k] rows (transposed tile)

  const int tid  = threadIdx.x;
  const int lane = tid & 31;
  const int wave = tid >> 5;
  const int bm = blockIdx.x * 128;
  const int bn = blockIdx.y * 128;
  const int wm = (wave & 3) * 32;
  const int wn = (wave >> 2) * 64;

  const int xrow = tid >> 1, xcs = (tid & 1) * 32;
  const int wrow = tid >> 2, wcs = (tid & 3) * 32;

  v8f acc[2][4] = {};

  for (int k0 = 0; k0 < D_MODEL; k0 += 64) {
    // stage X tile (128x64 f32 -> f16, row-major)
    {
      const float* src = x + (size_t)(bm + xrow) * D_MODEL + k0 + xcs;
      half_t* dst = &Xs[xrow * 72 + xcs];
#pragma unroll
      for (int i = 0; i < 32; ++i) dst[i] = (half_t)src[i];
      if (k0 + 64 < D_MODEL) __builtin_prefetch(src + 64, 0, 0);
    }
    // stage W tile (64x128 f32) transposed -> Wst[n][k]
    {
      const float* src = w + (size_t)(k0 + wrow) * N_QKV + bn + wcs;
#pragma unroll
      for (int i = 0; i < 32; ++i) Wst[(wcs + i) * 72 + wrow] = (half_t)src[i];
      if (k0 + 64 < D_MODEL) __builtin_prefetch(src + (size_t)64 * N_QKV, 0, 0);
    }
    __syncthreads();

#pragma unroll
    for (int kc = 0; kc < 64; kc += 32) {
      const v16h a0 = load_frag(&Xs[(wm + 0) * 72 + kc], 72, lane);
      const v16h a1 = load_frag(&Xs[(wm + 16) * 72 + kc], 72, lane);
#pragma unroll
      for (int i = 0; i < 4; ++i) {
        const v16h b = load_frag(&Wst[(wn + i * 16) * 72 + kc], 72, lane);
        acc[0][i] = wmma_f16(a0, b, acc[0][i]);
        acc[1][i] = wmma_f16(a1, b, acc[1][i]);
      }
    }
    __syncthreads();
  }

  // epilogue: bias add + scatter to q/k/v f16 buffers
  const int ncol = lane & 15;
  const int mhi  = (lane >> 4) * 8;
#pragma unroll
  for (int i = 0; i < 4; ++i) {
    const int gn = bn + wn + i * 16 + ncol;
    const float bv = bias[gn];
    const int which = gn >> 10;          // 0=q 1=k 2=v (uniform per block)
    const int hh = (gn & 1023) >> 6;
    const int hd = gn & 63;
    half_t* dstbuf = (which == 0) ? qf : ((which == 1) ? kf : vf);
#pragma unroll
    for (int ms = 0; ms < 2; ++ms)
#pragma unroll
      for (int r = 0; r < 8; ++r) {
        const int gm = bm + wm + ms * 16 + r + mhi;
        const int bb = gm >> 11;           // / SEQ
        const int tt = gm & (SEQ - 1);
        dstbuf[(size_t)(((bb * N_HEADS + hh) * SEQ) + tt) * HEAD_DIM + hd] =
            (half_t)(acc[ms][i][r] + bv);
      }
  }
}

// ---------------------------------------------------------------------------
// Kernel 2: causal flash attention. One block per (b*h, 128-query strip),
// one wave per 16-query tile. 32-key blocks; K staged by the Tensor Data
// Mover (with LDS padding -> stride 72), V staged transposed by the threads.
// ---------------------------------------------------------------------------
__global__ __launch_bounds__(256) void attn_kernel(
    const half_t* __restrict__ qf, const half_t* __restrict__ kf,
    const half_t* __restrict__ vf, half_t* __restrict__ of) {
  __shared__ half_t Ks[32 * 72];        // [key][hd], stride 72 (TDM-padded)
  __shared__ half_t Vst[64 * 40];       // [hd][key], stride 40
  __shared__ half_t Ps[8 * 16 * 40];    // per-wave P tile [m][k], stride 40

  const int tid  = threadIdx.x;
  const int lane = tid & 31;
  const int wave = tid >> 5;
  const int bh   = blockIdx.x;          // b*16 + h
  const int qblk = blockIdx.y;
  const int q0   = qblk * 128 + wave * 16;

  const half_t* qbase = qf + ((size_t)bh * SEQ + q0) * HEAD_DIM;
  const half_t* kbase = kf + (size_t)bh * SEQ * HEAD_DIM;
  const half_t* vbase = vf + (size_t)bh * SEQ * HEAD_DIM;
  const unsigned ksOff = (unsigned)(size_t)(void*)&Ks[0];

  const v16h qa0 = load_frag(qbase, HEAD_DIM, lane);       // hd 0..31
  const v16h qa1 = load_frag(qbase + 32, HEAD_DIM, lane);  // hd 32..63

  v8f o[4] = {};
  float mrow[8], lrow[8];
#pragma unroll
  for (int r = 0; r < 8; ++r) { mrow[r] = -1e30f; lrow[r] = 0.f; }

  const int nblocks = (qblk + 1) * 4;    // 32-key blocks, causal bound
  const float scale = 0.125f;            // 1/sqrt(64)
  const int ncol = lane & 15;
  const int mhi  = (lane >> 4) * 8;

  for (int kb = 0; kb < nblocks; ++kb) {
    const int kstart = kb * 32;
    // TDM: K block [32][64] -> Ks (stride 72), wave 0 only
    if (wave == 0) {
      tdm_load_k32x64(kbase + (size_t)kstart * HEAD_DIM, ksOff);
    }
    // cooperative stage of V (transposed)
    {
      const int row = tid >> 3;          // key 0..31
      const int cs  = (tid & 7) * 8;     // hd segment
      const v8h vv = *(const v8h*)(vbase + (size_t)(kstart + row) * HEAD_DIM + cs);
#pragma unroll
      for (int i = 0; i < 8; ++i) Vst[(cs + i) * 40 + row] = vv[i];
    }
    if (wave == 0) __builtin_amdgcn_s_wait_tensorcnt(0);
    __syncthreads();

    if (kstart <= q0 + 15) {             // wave-uniform causal skip
      v8f s0 = {}, s1 = {};
      {
        const v16h b0a = load_frag(&Ks[0], 72, lane);
        const v16h b0b = load_frag(&Ks[32], 72, lane);
        s0 = wmma_f16(qa0, b0a, s0);
        s0 = wmma_f16(qa1, b0b, s0);
        const v16h b1a = load_frag(&Ks[16 * 72], 72, lane);
        const v16h b1b = load_frag(&Ks[16 * 72 + 32], 72, lane);
        s1 = wmma_f16(qa0, b1a, s1);
        s1 = wmma_f16(qa1, b1b, s1);
      }
      float p0[8], p1[8], alpha[8];
#pragma unroll
      for (int r = 0; r < 8; ++r) {
        const int qrow = q0 + r + mhi;
        const float v0 = ((kstart + ncol) <= qrow) ? s0[r] * scale : -1e30f;
        const float v1 = ((kstart + 16 + ncol) <= qrow) ? s1[r] * scale : -1e30f;
        float rmax = fmaxf(v0, v1);
#pragma unroll
        for (int off = 1; off < 16; off <<= 1)
          rmax = fmaxf(rmax, __shfl_xor(rmax, off, 32));
        const float mn = fmaxf(mrow[r], rmax);
        const float a  = __expf(mrow[r] - mn);
        const float e0 = __expf(v0 - mn);
        const float e1 = __expf(v1 - mn);
        float rs = e0 + e1;
#pragma unroll
        for (int off = 1; off < 16; off <<= 1) rs += __shfl_xor(rs, off, 32);
        lrow[r] = lrow[r] * a + rs;
        mrow[r] = mn;
        alpha[r] = a;
        p0[r] = e0;
        p1[r] = e1;
      }
#pragma unroll
      for (int i = 0; i < 4; ++i)
#pragma unroll
        for (int r = 0; r < 8; ++r) o[i][r] *= alpha[r];

      // re-shape P: C-layout -> row-major LDS -> A fragment
      half_t* Pw = &Ps[wave * 16 * 40];
#pragma unroll
      for (int r = 0; r < 8; ++r) {
        const int m = r + mhi;
        Pw[m * 40 + ncol]      = (half_t)p0[r];
        Pw[m * 40 + 16 + ncol] = (half_t)p1[r];
      }
      const v16h pa = load_frag(Pw, 40, lane);  // same-wave LDS: in-order
#pragma unroll
      for (int i = 0; i < 4; ++i) {
        const v16h vb = load_frag(&Vst[(i * 16) * 40], 40, lane);
        o[i] = wmma_f16(pa, vb, o[i]);
      }
    }
    __syncthreads();
  }

  // epilogue: normalize and store attn output as f16 [b][t][d]
  const int bb = bh >> 4;
  const int hh = bh & 15;
#pragma unroll
  for (int i = 0; i < 4; ++i)
#pragma unroll
    for (int r = 0; r < 8; ++r) {
      const int qrow = q0 + r + mhi;
      const float val = o[i][r] / lrow[r];
      of[((size_t)bb * SEQ + qrow) * D_MODEL + hh * HEAD_DIM + i * 16 + ncol] =
          (half_t)val;
    }
}

// ---------------------------------------------------------------------------
// Kernel 3: output projection. out[8192,1024] = A_f16 @ o_w + o_b (f32 out)
// Same 128x128 / BK=64 tiling as kernel 1.
// ---------------------------------------------------------------------------
__global__ __launch_bounds__(256) void out_gemm_kernel(
    const half_t* __restrict__ a, const float* __restrict__ w,
    const float* __restrict__ bias, float* __restrict__ out) {
  __shared__ half_t Xs[128 * 72];
  __shared__ half_t Wst[128 * 72];

  const int tid  = threadIdx.x;
  const int lane = tid & 31;
  const int wave = tid >> 5;
  const int bm = blockIdx.x * 128;
  const int bn = blockIdx.y * 128;
  const int wm = (wave & 3) * 32;
  const int wn = (wave >> 2) * 64;

  const int xrow = tid >> 1, xcs = (tid & 1) * 32;
  const int wrow = tid >> 2, wcs = (tid & 3) * 32;

  v8f acc[2][4] = {};

  for (int k0 = 0; k0 < D_MODEL; k0 += 64) {
    // stage A tile (f16, straight copy)
    {
      const half_t* src = a + (size_t)(bm + xrow) * D_MODEL + k0 + xcs;
      half_t* dst = &Xs[xrow * 72 + xcs];
#pragma unroll
      for (int i = 0; i < 4; ++i)
        *(v8h*)(dst + i * 8) = *(const v8h*)(src + i * 8);
    }
    // stage W tile (64x128 f32) transposed
    {
      const float* src = w + (size_t)(k0 + wrow) * D_MODEL + bn + wcs;
#pragma unroll
      for (int i = 0; i < 32; ++i) Wst[(wcs + i) * 72 + wrow] = (half_t)src[i];
      if (k0 + 64 < D_MODEL) __builtin_prefetch(src + (size_t)64 * D_MODEL, 0, 0);
    }
    __syncthreads();

#pragma unroll
    for (int kc = 0; kc < 64; kc += 32) {
      const v16h a0 = load_frag(&Xs[(wm + 0) * 72 + kc], 72, lane);
      const v16h a1 = load_frag(&Xs[(wm + 16) * 72 + kc], 72, lane);
#pragma unroll
      for (int i = 0; i < 4; ++i) {
        const v16h b = load_frag(&Wst[(wn + i * 16) * 72 + kc], 72, lane);
        acc[0][i] = wmma_f16(a0, b, acc[0][i]);
        acc[1][i] = wmma_f16(a1, b, acc[1][i]);
      }
    }
    __syncthreads();
  }

  const int ncol = lane & 15;
  const int mhi  = (lane >> 4) * 8;
#pragma unroll
  for (int i = 0; i < 4; ++i) {
    const int gn = bn + wn + i * 16 + ncol;
    const float bv = bias[gn];
#pragma unroll
    for (int ms = 0; ms < 2; ++ms)
#pragma unroll
      for (int r = 0; r < 8; ++r) {
        const int gm = bm + wm + ms * 16 + r + mhi;
        out[(size_t)gm * D_MODEL + gn] = acc[ms][i][r] + bv;
      }
  }
}

extern "C" void kernel_launch(void* const* d_in, const int* in_sizes, int n_in,
                              void* d_out, int out_size, void* d_ws,
                              size_t ws_size, hipStream_t stream) {
  const float* x     = (const float*)d_in[0];
  const float* qkv_w = (const float*)d_in[1];
  const float* qkv_b = (const float*)d_in[2];
  const float* o_w   = (const float*)d_in[3];
  const float* o_b   = (const float*)d_in[4];
  float* out = (float*)d_out;

  // workspace: 4 f16 buffers of 8M elements each = 64 MB
  half_t* qf = (half_t*)d_ws;
  half_t* kf = qf + (size_t)MROWS * D_MODEL;
  half_t* vf = kf + (size_t)MROWS * D_MODEL;
  half_t* af = vf + (size_t)MROWS * D_MODEL;

  dim3 blk(256, 1, 1);

  dim3 g1(MROWS / 128, N_QKV / 128, 1);          // 64 x 24
  qkv_gemm_kernel<<<g1, blk, 0, stream>>>(x, qkv_w, qkv_b, qf, kf, vf);

  dim3 g2(BATCH * N_HEADS, SEQ / 128, 1);        // 64 x 16
  attn_kernel<<<g2, blk, 0, stream>>>(qf, kf, vf, af);

  dim3 g3(MROWS / 128, D_MODEL / 128, 1);        // 64 x 8
  out_gemm_kernel<<<g3, blk, 0, stream>>>(af, o_w, o_b, out);
}
